// MPNEncoder_72816875537080
// MI455X (gfx1250) — matrix operand order repaired and checked by reading, not compiled, not run on hardware
//
#include <hip/hip_runtime.h>

// ============================================================================
// Chemprop bond-message MPNN for MI455X (gfx1250), bf16 WMMA + fp32 accumulate
//
// GEMMs use v_wmma_f32_16x16x32_bf16 with:
//   - async global->LDS staging of the B k-slice (global_load_async_to_lds_b128,
//     ASYNCcnt + s_wait_asynccnt), double-buffered in LDS (2 x 25.6KB),
//   - __builtin_amdgcn_sched_group_barrier pipelining: DS-read groups issued
//     one WMMA-group ahead (batched loads, partial DScnt waits, 4+ WMMAs
//     back-to-back per group),
//   - A-fragment prefetch across k-steps,
//   - 80B LDS row pitch -> conflict-free ds_load_b128 B-fragment reads.
// Activations kept bf16 so the 200001x320 message matrix (128MB) stays
// L2-resident (192MB) for the neighbor gathers. Workspace ~577MB with aliasing.
// ============================================================================

#ifndef USE_ASYNC_STAGE
#define USE_ASYNC_STAGE 1
#endif

typedef __bf16 bf16_t;
typedef __bf16 v16bf __attribute__((ext_vector_type(16)));
typedef float  v8f   __attribute__((ext_vector_type(8)));

#define NMOLS  2000
#define APM    50
#define NATOMS 100001
#define NBONDS 200001
#define MAXNB  6
#define AFD    133
#define BFD    147
#define HID    300
#define NPAD   320      // padded N for all GEMMs (20 tiles of 16)
#define KPI    160      // padded K for f_bonds GEMM (5 ksteps of 32)
#define KPH    320      // padded K for W_h GEMM (10 ksteps)
#define KPO    480      // padded K for W_o GEMM: [a_msg 0..319 | f_atoms 320..452 | 0]
#define LDSROW 5        // uint4 per LDS row (80B pitch, 64B payload) -> bank-conflict-free

// sched_group_barrier masks (AMDGPU backend)
#define SGB_MFMA_WMMA 0x008
#define SGB_DS_READ   0x100

union Frag32 { uint4 u[2]; v16bf v; };

// ---------------------------------------------------------------------------
// Weight packing: out[n][k] = bf16(W[k][n]) padded with zeros. N is HID->NPAD.
// ---------------------------------------------------------------------------
__global__ void pack_wT(const float* __restrict__ W, int K, int Kpad,
                        bf16_t* __restrict__ out) {
  long gid = (long)blockIdx.x * blockDim.x + threadIdx.x;
  long tot = (long)NPAD * Kpad;
  if (gid >= tot) return;
  int n = (int)(gid / Kpad), k = (int)(gid % Kpad);
  float v = (n < HID && k < K) ? W[(long)k * HID + n] : 0.0f;
  out[gid] = (bf16_t)v;
}

// W_o pack with reordered K: k<300 -> W_o[133+k][n] (a_message part),
// 320<=k<453 -> W_o[k-320][n] (f_atoms part), else 0.
__global__ void pack_wo(const float* __restrict__ Wo, bf16_t* __restrict__ out) {
  long gid = (long)blockIdx.x * blockDim.x + threadIdx.x;
  long tot = (long)NPAD * KPO;
  if (gid >= tot) return;
  int n = (int)(gid / KPO), k = (int)(gid % KPO);
  float v = 0.0f;
  if (n < HID) {
    if (k < HID)                          v = Wo[(long)(AFD + k) * HID + n];
    else if (k >= NPAD && k < NPAD + AFD) v = Wo[(long)(k - NPAD) * HID + n];
  }
  out[gid] = (bf16_t)v;
}

// f_bonds [NBONDS][147] fp32 -> bf16 A-buffer [NBONDS][160], zero padded
__global__ void pack_bonds(const float* __restrict__ fb, bf16_t* __restrict__ out) {
  long gid = (long)blockIdx.x * blockDim.x + threadIdx.x;
  long tot = (long)NBONDS * KPI;
  if (gid >= tot) return;
  int b = (int)(gid / KPI), k = (int)(gid % KPI);
  float v = (k < BFD) ? fb[(long)b * BFD + k] : 0.0f;
  out[gid] = (bf16_t)v;
}

// f_atoms [NATOMS][133] fp32 -> columns [320..479] of Afin [NATOMS][480]
__global__ void pack_fatoms(const float* __restrict__ fa, bf16_t* __restrict__ afin) {
  long gid = (long)blockIdx.x * blockDim.x + threadIdx.x;
  long tot = (long)NATOMS * (KPO - NPAD);   // 160 per atom
  if (gid >= tot) return;
  int a  = (int)(gid / (KPO - NPAD));
  int kk = (int)(gid % (KPO - NPAD));
  float v = (kk < AFD) ? fa[(long)a * AFD + kk] : 0.0f;
  afin[(long)a * KPO + NPAD + kk] = (bf16_t)v;
}

// ---------------------------------------------------------------------------
// Neighbor aggregation: out[a][k] = sum_j w_bonds[a2b[a][j]] * msg[a2b[a][j]][k]
// ---------------------------------------------------------------------------
__global__ void agg_kernel(const bf16_t* __restrict__ msg,
                           const int* __restrict__ a2b,
                           const float* __restrict__ w_bonds,
                           bf16_t* __restrict__ out, int out_stride) {
  long gid = (long)blockIdx.x * blockDim.x + threadIdx.x;
  long tot = (long)NATOMS * (NPAD / 8);
  if (gid >= tot) return;
  int a  = (int)(gid / (NPAD / 8));
  int kc = (int)(gid % (NPAD / 8)) * 8;
  const int* nb = a2b + (long)a * MAXNB;
  float acc[8];
#pragma unroll
  for (int e = 0; e < 8; ++e) acc[e] = 0.0f;
#pragma unroll
  for (int j = 0; j < MAXNB; ++j) {
    int b = nb[j];
    float w = w_bonds[b];
    uint4 q = *(const uint4*)(msg + (long)b * NPAD + kc);
    const bf16_t* p = (const bf16_t*)&q;
#pragma unroll
    for (int e = 0; e < 8; ++e) acc[e] += w * (float)p[e];
  }
  uint4 r;
  bf16_t* rp = (bf16_t*)&r;
#pragma unroll
  for (int e = 0; e < 8; ++e) rp[e] = (bf16_t)acc[e];
  *(uint4*)(out + (long)a * out_stride + kc) = r;
}

// ---------------------------------------------------------------------------
// m2[b][k] = a_msg[b2a[b]][k] - w_bonds[b] * msg[b2revb[b]][k]  (bf16 out)
// ---------------------------------------------------------------------------
__global__ void m2_kernel(const bf16_t* __restrict__ amsg,
                          const bf16_t* __restrict__ msg,
                          const int* __restrict__ b2a,
                          const int* __restrict__ b2revb,
                          const float* __restrict__ w_bonds,
                          bf16_t* __restrict__ out) {
  long gid = (long)blockIdx.x * blockDim.x + threadIdx.x;
  long tot = (long)NBONDS * (NPAD / 8);
  if (gid >= tot) return;
  int b  = (int)(gid / (NPAD / 8));
  int kc = (int)(gid % (NPAD / 8)) * 8;
  int ia = b2a[b], ir = b2revb[b];
  float wb = w_bonds[b];
  uint4 qa = *(const uint4*)(amsg + (long)ia * NPAD + kc);
  uint4 qr = *(const uint4*)(msg  + (long)ir * NPAD + kc);
  const bf16_t* pa = (const bf16_t*)&qa;
  const bf16_t* pr = (const bf16_t*)&qr;
  uint4 r;
  bf16_t* rp = (bf16_t*)&r;
#pragma unroll
  for (int e = 0; e < 8; ++e)
    rp[e] = (bf16_t)((float)pa[e] - wb * (float)pr[e]);
  *(uint4*)(out + (long)b * NPAD + kc) = r;
}

// ---------------------------------------------------------------------------
// bf16 WMMA GEMM: C[M][320] = A[M][lda] @ Bp^T  (Bp packed [320][ldb], ldb=Kpad)
// 256 threads = 8 waves; wave w owns rows [128*bx + 16w .. +15] and all 20
// N-tiles (20 x v8f accumulators).
// ---------------------------------------------------------------------------
__global__ void __launch_bounds__(256, 1)
gemm_bf16_wmma(const bf16_t* __restrict__ A, int lda,
               const bf16_t* __restrict__ Bp, int ldb, int ksteps,
               const float* __restrict__ addf, int ldadd,
               const float* __restrict__ bias,
               float* __restrict__ outF, int ldoF,
               bf16_t* __restrict__ outH, int ldoH,
               int M) {
  __shared__ uint4 Bs[2][NPAD * LDSROW];             // 2 x 25.6 KB, 80B row pitch

  const int tid  = threadIdx.x;
  const int wave = tid >> 5;
  const int lane = tid & 31;
  const int half = lane >> 4;                        // 0: K 0..7/16..23, 1: K 8..15/24..31
  const int lrow = lane & 15;
  const int rowBase = blockIdx.x * 128 + wave * 16;

  int arow = rowBase + lrow;
  if (arow >= M) arow = M - 1;                       // clamp for loads; stores guarded
  const char* aBase = (const char*)A + (size_t)arow * lda * 2;
  const size_t ldb2 = (size_t)ldb * 2;

  const unsigned ldsBase0 = (unsigned)(uintptr_t)(void*)&Bs[0][0];

  // Stage B[:, ks*32 .. +31] (320 rows x 64B) into LDS buffer `buf`.
  auto stage = [&](int ks, int buf) {
#if USE_ASYNC_STAGE
    unsigned long long sbase = (unsigned long long)(uintptr_t)Bp + (unsigned long long)ks * 64ull;
    for (int c = tid; c < NPAD * 4; c += 256) {
      int n = c >> 2, p = c & 3;
      unsigned voff  = (unsigned)((size_t)n * ldb2 + (size_t)p * 16);
      unsigned laddr = ldsBase0 + (unsigned)(buf * NPAD * LDSROW * 16 + (n * LDSROW + p) * 16);
      asm volatile("global_load_async_to_lds_b128 %0, %1, %2"
                   :: "v"(laddr), "v"(voff), "s"(sbase) : "memory");
    }
#else
    for (int c = tid; c < NPAD * 4; c += 256) {
      int n = c >> 2, p = c & 3;
      Bs[buf][n * LDSROW + p] =
          *(const uint4*)((const char*)Bp + (size_t)n * ldb2 + (size_t)ks * 64 + p * 16);
    }
#endif
  };

  v8f acc[20];
#pragma unroll
  for (int t = 0; t < 20; ++t) acc[t] = (v8f)0.0f;

  // Prologue: stage k-slice 0, load A fragment 0.
  stage(0, 0);
  Frag32 af_cur;
  af_cur.u[0] = *(const uint4*)(aBase + half * 16);
  af_cur.u[1] = *(const uint4*)(aBase + half * 16 + 32);

  for (int ks = 0; ks < ksteps; ++ks) {
#if USE_ASYNC_STAGE
    asm volatile("s_wait_asynccnt 0" ::: "memory");  // this wave's staged chunks landed
#endif
    __syncthreads();                                 // whole block's slice ready

    if (ks + 1 < ksteps) stage(ks + 1, (ks + 1) & 1);  // overlap next-slice DMA

    // Prefetch A fragment for next k-step (used next iteration).
    Frag32 af_next = af_cur;
    if (ks + 1 < ksteps) {
      af_next.u[0] = *(const uint4*)(aBase + (ks + 1) * 64 + half * 16);
      af_next.u[1] = *(const uint4*)(aBase + (ks + 1) * 64 + half * 16 + 32);
    }

    const uint4* bufp = &Bs[ks & 1][0];
#pragma unroll
    for (int t = 0; t < 20; ++t) {
      Frag32 bf;
      const uint4* bs = &bufp[(t * 16 + lrow) * LDSROW + half];
      bf.u[0] = bs[0];
      bf.u[1] = bs[2];
      acc[t] = __builtin_amdgcn_wmma_f32_16x16x32_bf16(
          false, af_cur.v, false, bf.v, (short)0, acc[t], false, false);
    }

    // Schedule shaping (40 ds_reads, 20 WMMAs per k-step): keep DS-read groups
    // one WMMA-group ahead -> batched loads, partial DScnt waits, WMMAs issue
    // back-to-back. Soft directives: ignored if infeasible.
    __builtin_amdgcn_sched_group_barrier(SGB_DS_READ, 16, 0);   // tiles 0-7 loads
#pragma unroll
    for (int g = 0; g < 3; ++g) {
      __builtin_amdgcn_sched_group_barrier(SGB_MFMA_WMMA, 4, 0); // wmma group g
      __builtin_amdgcn_sched_group_barrier(SGB_DS_READ, 8, 0);   // loads group g+2
    }
    __builtin_amdgcn_sched_group_barrier(SGB_MFMA_WMMA, 8, 0);   // wmma groups 3,4

    af_cur = af_next;
  }

  // Epilogue: C layout — VGPR r: lanes0-15 row=r, lanes16-31 row=r+8; col=lane%16
#pragma unroll
  for (int t = 0; t < 20; ++t) {
    int col = t * 16 + lrow;
    float bval = 0.0f;
    if (bias) bval = (col < HID) ? bias[col] : 0.0f;
#pragma unroll
    for (int r = 0; r < 8; ++r) {
      int row = rowBase + r + half * 8;
      if (row < M) {
        float v = acc[t][r] + bval;
        if (addf) v += addf[(size_t)row * ldadd + col];
        if (outF) outF[(size_t)row * ldoF + col] = v;
        if (outH) {
          float rv = v > 0.0f ? v : 0.0f;
          outH[(size_t)row * ldoH + col] = (bf16_t)rv;
        }
      }
    }
  }
}

// ---------------------------------------------------------------------------
// Per-molecule weighted-mean readout. setup_inputs builds mol m from the 50
// contiguous atoms [1+50m, 50+50m] (atom 0 = padding -> dropped segment).
// ---------------------------------------------------------------------------
__global__ void readout_kernel(const bf16_t* __restrict__ ah,       // [NATOMS][320]
                               const float* __restrict__ w_atoms,
                               const float* __restrict__ deg,
                               float* __restrict__ out) {            // [NMOLS][300]
  int m = blockIdx.x;
  int c = threadIdx.x;          // 0..319
  float acc = 0.0f, wsum = 0.0f;
  int base = 1 + m * APM;
  for (int i = 0; i < APM; ++i) {
    int a = base + i;
    float w = w_atoms[a];
    wsum += w;
    acc += w * (float)ah[(long)a * NPAD + c];
  }
  if (c < HID) out[(long)m * HID + c] = deg[m] * (acc / wsum);
}

// ===========================================================================
extern "C" void kernel_launch(void* const* d_in, const int* in_sizes, int n_in,
                              void* d_out, int out_size, void* d_ws, size_t ws_size,
                              hipStream_t stream) {
  const float* f_atoms = (const float*)d_in[0];
  const float* f_bonds = (const float*)d_in[1];
  const float* w_atoms = (const float*)d_in[2];
  const float* w_bonds = (const float*)d_in[3];
  const float* deg     = (const float*)d_in[4];
  const float* W_i     = (const float*)d_in[5];
  const float* W_h     = (const float*)d_in[6];
  const float* W_o     = (const float*)d_in[7];
  const float* b_o     = (const float*)d_in[8];
  const int*   a2b     = (const int*)d_in[9];
  const int*   b2a     = (const int*)d_in[10];
  const int*   b2revb  = (const int*)d_in[11];
  float* out = (float*)d_out;
  (void)in_sizes; (void)n_in; (void)out_size; (void)ws_size;

  // ---- workspace layout (256B aligned) ----
  char* ws = (char*)d_ws;
  auto align256 = [](size_t x) { return (x + 255) & ~(size_t)255; };
  size_t off = 0;
  bf16_t* Wi_p = (bf16_t*)(ws + off); off = align256(off + (size_t)NPAD * KPI * 2);
  bf16_t* Wh_p = (bf16_t*)(ws + off); off = align256(off + (size_t)NPAD * KPH * 2);
  bf16_t* Wo_p = (bf16_t*)(ws + off); off = align256(off + (size_t)NPAD * KPO * 2);
  size_t off_inp = off;                         // fp32 inp [NBONDS][320]; later
  float*  inp  = (float*)(ws + off_inp);        //   aliased by atom_hiddens bf16
  off = align256(off_inp + (size_t)NBONDS * NPAD * 4);
  bf16_t* message = (bf16_t*)(ws + off); off = align256(off + (size_t)NBONDS * NPAD * 2);
  bf16_t* Abuf    = (bf16_t*)(ws + off); off = align256(off + (size_t)NBONDS * NPAD * 2); // A0/m2/Afin shared
  bf16_t* amsg    = (bf16_t*)(ws + off); off = align256(off + (size_t)NATOMS * NPAD * 2);
  bf16_t* atomh   = (bf16_t*)(ws + off_inp);    // alias: inp dead after loop GEMMs

  const int T = 256;
  auto nb = [](long work, int t) { return (unsigned)((work + t - 1) / t); };

  // ---- weight + input packing ----
  pack_wT<<<nb((long)NPAD * KPI, T), T, 0, stream>>>(W_i, BFD, KPI, Wi_p);
  pack_wT<<<nb((long)NPAD * KPH, T), T, 0, stream>>>(W_h, HID, KPH, Wh_p);
  pack_wo<<<nb((long)NPAD * KPO, T), T, 0, stream>>>(W_o, Wo_p);
  pack_bonds<<<nb((long)NBONDS * KPI, T), T, 0, stream>>>(f_bonds, Abuf);

  // ---- inp = f_bonds @ W_i (fp32, raw) ; message = relu(inp) (bf16) ----
  gemm_bf16_wmma<<<nb(NBONDS, 128), T, 0, stream>>>(
      Abuf, KPI, Wi_p, KPI, KPI / 32,
      nullptr, 0, nullptr, inp, NPAD, message, NPAD, NBONDS);

  // ---- DEPTH-1 message-passing iterations ----
  for (int it = 0; it < 3; ++it) {
    agg_kernel<<<nb((long)NATOMS * (NPAD / 8), T), T, 0, stream>>>(
        message, a2b, w_bonds, amsg, NPAD);
    m2_kernel<<<nb((long)NBONDS * (NPAD / 8), T), T, 0, stream>>>(
        amsg, message, b2a, b2revb, w_bonds, Abuf);
    gemm_bf16_wmma<<<nb(NBONDS, 128), T, 0, stream>>>(
        Abuf, KPH, Wh_p, KPH, KPH / 32,
        inp, NPAD, nullptr, nullptr, 0, message, NPAD, NBONDS);
  }

  // ---- final aggregation into concat A-buffer + f_atoms pack ----
  agg_kernel<<<nb((long)NATOMS * (NPAD / 8), T), T, 0, stream>>>(
      message, a2b, w_bonds, Abuf, KPO);            // cols 0..319 of Afin
  pack_fatoms<<<nb((long)NATOMS * (KPO - NPAD), T), T, 0, stream>>>(f_atoms, Abuf);

  // ---- atom_hiddens = relu(a_input @ W_o + b_o) (bf16, aliases inp) ----
  gemm_bf16_wmma<<<nb(NATOMS, 128), T, 0, stream>>>(
      Abuf, KPO, Wo_p, KPO, KPO / 32,
      nullptr, 0, b_o, nullptr, 0, atomh, NPAD, NATOMS);

  // ---- per-molecule readout ----
  readout_kernel<<<NMOLS, NPAD, 0, stream>>>(atomh, w_atoms, deg, out);
}
